// sLSTMBlock_50551765074672
// MI455X (gfx1250) — matrix operand order repaired and checked by reading, not compiled, not run on hardware
//
#include <hip/hip_runtime.h>
#include <hip/hip_bf16.h>
#include <math.h>

// ---------------------------------------------------------------------------
// sLSTM block for MI455X (gfx1250, wave32, WMMA).
//   B=4, T=2048, C=1024, H=4, D=256
// Pipeline:
//   k_cvt      : f32 -> bf16 weight conversion (wx_w, out_w)
//   k_conv     : depthwise causal conv(k=4) + SiLU -> xc (bf16)
//   k_gemm     : gates = xc @ wx_w^T + wx_b, tanh on z columns (WMMA bf16,
//                double-buffered LDS, async global->LDS when available)
//   k_scan     : sequential sLSTM recurrence over T, fused GroupNorm stats
//   k_gnapply  : (h-mean)*rstd*gn_w+gn_b -> bf16
//   k_gemm     : out = hn @ out_w^T + out_b + x (WMMA bf16, fused residual)
// ---------------------------------------------------------------------------

typedef __attribute__((ext_vector_type(16))) __bf16 v16bf;
typedef __attribute__((ext_vector_type(8)))  __bf16 v8bf;
typedef __attribute__((ext_vector_type(8)))  float  v8f;

#define Bv 4
#define Tv 2048
#define Cv 1024
#define Hh 4
#define Dd 256
#define Mrows (Bv * Tv)           // 8192
#define Ngates (4 * Cv)           // 4096

// ---- CDNA5 async global->LDS path (GLOBAL_LOAD_ASYNC_TO_LDS_B128) ---------
#if defined(__has_builtin)
#if __has_builtin(__builtin_amdgcn_global_load_async_to_lds_b128) && \
    __has_builtin(__builtin_amdgcn_s_wait_asynccnt)
#define USE_ASYNC_LDS 1
#endif
#endif
#ifndef USE_ASYNC_LDS
#define USE_ASYNC_LDS 0
#endif

#if USE_ASYNC_LDS
// builtin parameter type per hipcc diagnostic:
//   '__attribute__((__vector_size__(4 * sizeof(int)))) int __device__ *'
typedef int v4i_g __attribute__((vector_size(4 * sizeof(int))));
typedef __attribute__((address_space(1))) v4i_g as1_v4i;
typedef __attribute__((address_space(3))) v4i_g as3_v4i;
__device__ __forceinline__ void async_cp_b128(const void* g, void* l) {
  // generic LDS VA low 32 bits == LDS byte offset (ISA flat->LDS mapping)
  __builtin_amdgcn_global_load_async_to_lds_b128(
      (as1_v4i*)(uintptr_t)g, (as3_v4i*)(unsigned)(uintptr_t)l, 0, 0);
}
#endif

// inf-safe fast tanh via hardware exp: sign(x)*(1 - 2/(e^{2|x|}+1))
__device__ __forceinline__ float fast_tanhf(float v) {
  float e = __expf(2.0f * fabsf(v));
  return __builtin_copysignf(1.0f - 2.0f / (e + 1.0f), v);
}

// ------------------------------ f32 -> bf16 --------------------------------
__global__ __launch_bounds__(256) void k_cvt(const float* __restrict__ in,
                                             __bf16* __restrict__ out,
                                             long n) {
  long i = (long)blockIdx.x * 256 + threadIdx.x;
  if (i < n) out[i] = (__bf16)in[i];
}

// --------------------- depthwise conv + SiLU -> bf16 -----------------------
__global__ __launch_bounds__(256) void k_conv(const float* __restrict__ x,
                                              const float* __restrict__ cw,  // (C,1,4)
                                              const float* __restrict__ cb,  // (C)
                                              __bf16* __restrict__ xcb) {
  long i = (long)blockIdx.x * 256 + threadIdx.x;   // over B*T*C
  if (i >= (long)Mrows * Cv) return;
  int c = (int)(i % Cv);
  long bt = i / Cv;
  int t = (int)(bt % Tv);
  long brow = bt - t;                              // b*T
  float acc = cb[c];
  #pragma unroll
  for (int j = 0; j < 4; ++j) {
    int tt = t - 3 + j;
    if (tt >= 0) acc = fmaf(x[(brow + tt) * (long)Cv + c], cw[c * 4 + j], acc);
  }
  float s = acc / (1.0f + __expf(-acc));           // SiLU
  xcb[i] = (__bf16)s;
}

// --------------------------- WMMA GEMM kernel ------------------------------
// out[M,N] = A[M,K] @ W[N,K]^T + bias[N]  (optional tanh on z-gate cols,
//                                          optional residual add)
// Block: 256 thr = 8 waves. Block tile 128x128, K-step 32, double-buffered.
// Wave grid 4(M) x 2(N): each wave -> 32x64 = 2x4 tiles of 16x16.
template <int TANH_Z, int RESIDUAL>
__global__ __launch_bounds__(256) void k_gemm(const __bf16* __restrict__ A,
                                              const __bf16* __restrict__ W,
                                              const float* __restrict__ bias,
                                              const float* __restrict__ resid,
                                              float* __restrict__ out,
                                              int M, int N, int K) {
  constexpr int BK  = 32;
  constexpr int LDA = BK + 8;  // 40 halves = 80 B row stride (conflict-free)
  __shared__ __attribute__((aligned(16))) __bf16 As[2][128 * LDA];
  __shared__ __attribute__((aligned(16))) __bf16 Bs[2][128 * LDA];

  const int tid   = threadIdx.x;
  const int lane  = tid & 31;
  const int wave  = tid >> 5;
  const int wm    = wave & 3;                      // 4 waves along M
  const int wn    = wave >> 2;                     // 2 waves along N
  const int lhalf = lane & 15;                     // 0..15
  const int lhi   = lane >> 4;                     // 0/1
  const int m0    = blockIdx.y * 128;
  const int n0    = blockIdx.x * 128;

  // per-thread staging coords: 256 thr x 8 halves x 2 passes per 128x32 tile
  const int row0 = tid >> 2;          // pass 0 row
  const int ch0  = tid & 3;
  const int row1 = (tid + 256) >> 2;  // pass 1 row
  const int ch1  = (tid + 256) & 3;

  v8f acc[2][4];
  #pragma unroll
  for (int mt = 0; mt < 2; ++mt)
    #pragma unroll
    for (int nt = 0; nt < 4; ++nt) acc[mt][nt] = (v8f){};

  const int nk = K / BK;

#if USE_ASYNC_LDS
  // ---------- async global->LDS double-buffered pipeline ----------
  auto issue_copy = [&](int buf, int k0) {
    async_cp_b128(A + (long)(m0 + row0) * K + k0 + ch0 * 8,
                  &As[buf][row0 * LDA + ch0 * 8]);
    async_cp_b128(W + (long)(n0 + row0) * K + k0 + ch0 * 8,
                  &Bs[buf][row0 * LDA + ch0 * 8]);
    async_cp_b128(A + (long)(m0 + row1) * K + k0 + ch1 * 8,
                  &As[buf][row1 * LDA + ch1 * 8]);
    async_cp_b128(W + (long)(n0 + row1) * K + k0 + ch1 * 8,
                  &Bs[buf][row1 * LDA + ch1 * 8]);
  };
  issue_copy(0, 0);
#else
  // ---------- register-staged double-buffered pipeline ----------
  v8bf rga[2], rgb[2];
  auto load_regs = [&](int k0) {
    rga[0] = *(const v8bf*)(A + (long)(m0 + row0) * K + k0 + ch0 * 8);
    rgb[0] = *(const v8bf*)(W + (long)(n0 + row0) * K + k0 + ch0 * 8);
    rga[1] = *(const v8bf*)(A + (long)(m0 + row1) * K + k0 + ch1 * 8);
    rgb[1] = *(const v8bf*)(W + (long)(n0 + row1) * K + k0 + ch1 * 8);
  };
  auto store_lds = [&](int buf) {
    *(v8bf*)(&As[buf][row0 * LDA + ch0 * 8]) = rga[0];
    *(v8bf*)(&Bs[buf][row0 * LDA + ch0 * 8]) = rgb[0];
    *(v8bf*)(&As[buf][row1 * LDA + ch1 * 8]) = rga[1];
    *(v8bf*)(&Bs[buf][row1 * LDA + ch1 * 8]) = rgb[1];
  };
  load_regs(0);
  store_lds(0);
  __syncthreads();
#endif

  for (int kk = 0; kk < nk; ++kk) {
    const int cur = kk & 1;

#if USE_ASYNC_LDS
    if (kk + 1 < nk) {
      issue_copy(cur ^ 1, (kk + 1) * BK);
      // <=4 outstanding == only next tile's 4 async loads; current complete
      __builtin_amdgcn_s_wait_asynccnt(4);
    } else {
      __builtin_amdgcn_s_wait_asynccnt(0);
    }
    __syncthreads();
#else
    if (kk + 1 < nk) load_regs((kk + 1) * BK);  // overlap with compute below
#endif

    // A fragments: lane L holds row m=L%16; K-chunks {g*8..} and {16+g*8..}.
    v16bf afrag[2];
    #pragma unroll
    for (int mt = 0; mt < 2; ++mt) {
      const __bf16* base = &As[cur][(wm * 32 + mt * 16 + lhalf) * LDA];
      v8bf lo = *(const v8bf*)(base + lhi * 8);
      v8bf hi = *(const v8bf*)(base + 16 + lhi * 8);
      afrag[mt] = __builtin_shufflevector(lo, hi, 0, 1, 2, 3, 4, 5, 6, 7, 8, 9,
                                          10, 11, 12, 13, 14, 15);
    }
    // B fragments: lane L holds col n=L%16; K-chunks {g*8..} and {16+g*8..}.
    v16bf bfrag[4];
    #pragma unroll
    for (int nt = 0; nt < 4; ++nt) {
      const __bf16* base = &Bs[cur][(wn * 64 + nt * 16 + lhalf) * LDA];
      v8bf lo = *(const v8bf*)(base + lhi * 8);
      v8bf hi = *(const v8bf*)(base + 16 + lhi * 8);
      bfrag[nt] = __builtin_shufflevector(lo, hi, 0, 1, 2, 3, 4, 5, 6, 7, 8, 9,
                                          10, 11, 12, 13, 14, 15);
    }
    #pragma unroll
    for (int mt = 0; mt < 2; ++mt)
      #pragma unroll
      for (int nt = 0; nt < 4; ++nt)
        acc[mt][nt] = __builtin_amdgcn_wmma_f32_16x16x32_bf16(
            false, afrag[mt], false, bfrag[nt], (short)0, acc[mt][nt], false,
            false);

    __syncthreads();   // all waves done reading buf `cur` before it is refilled

#if !USE_ASYNC_LDS
    if (kk + 1 < nk) {
      store_lds(cur ^ 1);
      __syncthreads();
    }
#endif
  }

  // Epilogue: C layout — VGPR r: lanes 0-15 -> M=r, lanes 16-31 -> M=8+r.
  #pragma unroll
  for (int mt = 0; mt < 2; ++mt) {
    #pragma unroll
    for (int nt = 0; nt < 4; ++nt) {
      int n = n0 + wn * 64 + nt * 16 + lhalf;
      float bn = bias[n];
      #pragma unroll
      for (int r = 0; r < 8; ++r) {
        int m = m0 + wm * 32 + mt * 16 + lhi * 8 + r;
        float v = acc[mt][nt][r] + bn;
        if (TANH_Z) {
          if ((n >> 10) == 2) v = fast_tanhf(v);   // z-gate columns [2C,3C)
        }
        if (RESIDUAL) v += resid[(long)m * N + n];
        out[(long)m * N + n] = v;
      }
    }
  }
}

// -------------------- sLSTM scan + GroupNorm statistics --------------------
// One block per (b,h): 256 threads = D channels; serial over T.
__global__ __launch_bounds__(256) void k_scan(const float* __restrict__ gates, // (B*T, 4C)
                                              float* __restrict__ h,           // (B*T, C)
                                              float* __restrict__ stats) {     // (B*H, 2)
  const int d  = threadIdx.x;
  const int bh = blockIdx.x;
  const int b  = bh >> 2;
  const int hh = bh & 3;

  float c = 0.f, n = 1.f, m = 0.f, sum = 0.f, sq = 0.f;
  for (int t = 0; t < Tv; ++t) {
    long base = ((long)(b * Tv + t)) * Ngates + hh * Dd + d;
    float ig = gates[base + 0 * Cv];
    float fg = gates[base + 1 * Cv];
    float zg = gates[base + 2 * Cv];                // already tanh'd
    float og = gates[base + 3 * Cv];
    float mt = fmaxf(fg + m, ig);
    float ip = __expf(ig - mt);
    float fp = __expf(fg + m - mt);
    c = fp * c + ip * zg;
    n = fp * n + ip;
    m = mt;
    float hv = (1.0f / (1.0f + __expf(-og))) * (c / (n + 1e-6f));
    h[((long)(b * Tv + t)) * Cv + hh * Dd + d] = hv;
    sum += hv;
    sq += hv * hv;
  }

  __shared__ float rs[256], rq[256];
  rs[d] = sum;
  rq[d] = sq;
  __syncthreads();
  for (int s = 128; s > 0; s >>= 1) {
    if (d < s) { rs[d] += rs[d + s]; rq[d] += rq[d + s]; }
    __syncthreads();
  }
  if (d == 0) {
    float cnt  = (float)Tv * (float)Dd;
    float mean = rs[0] / cnt;
    float var  = rq[0] / cnt - mean * mean;
    stats[bh * 2 + 0] = mean;
    stats[bh * 2 + 1] = rsqrtf(var + 1e-5f);
  }
}

// --------------------------- GroupNorm apply -------------------------------
__global__ __launch_bounds__(256) void k_gnapply(const float* __restrict__ h,
                                                 const float* __restrict__ stats,
                                                 const float* __restrict__ gw,
                                                 const float* __restrict__ gb,
                                                 __bf16* __restrict__ hnb) {
  long i = (long)blockIdx.x * 256 + threadIdx.x;   // over B*T*C
  if (i >= (long)Mrows * Cv) return;
  int c = (int)(i % Cv);
  int b = (int)(i / ((long)Tv * Cv));
  int hh = c >> 8;
  float mean = stats[(b * 4 + hh) * 2 + 0];
  float rstd = stats[(b * 4 + hh) * 2 + 1];
  float v = (h[i] - mean) * rstd * gw[c] + gb[c];
  hnb[i] = (__bf16)v;
}

// ------------------------------- launcher ----------------------------------
extern "C" void kernel_launch(void* const* d_in, const int* in_sizes, int n_in,
                              void* d_out, int out_size, void* d_ws,
                              size_t ws_size, hipStream_t stream) {
  const float* x      = (const float*)d_in[0];
  const float* conv_w = (const float*)d_in[1];
  const float* conv_b = (const float*)d_in[2];
  const float* wx_w   = (const float*)d_in[3];
  const float* wx_b   = (const float*)d_in[4];
  const float* gn_w   = (const float*)d_in[5];
  const float* gn_b   = (const float*)d_in[6];
  const float* out_w  = (const float*)d_in[7];
  const float* out_b  = (const float*)d_in[8];
  float* out = (float*)d_out;

  // workspace carve-up (256B aligned)
  char* ws = (char*)d_ws;
  size_t off = 0;
  auto alloc = [&](size_t bytes) {
    char* p = ws + off;
    off += (bytes + 255) & ~(size_t)255;
    return p;
  };
  __bf16* xcb    = (__bf16*)alloc((size_t)Mrows * Cv * 2);       // 16 MB
  __bf16* wxw_b  = (__bf16*)alloc((size_t)Ngates * Cv * 2);      //  8 MB
  __bf16* outw_b = (__bf16*)alloc((size_t)Cv * Cv * 2);          //  2 MB
  float*  gates  = (float*)alloc((size_t)Mrows * Ngates * 4);    // 128 MB
  float*  hbuf   = (float*)alloc((size_t)Mrows * Cv * 4);        // 32 MB
  __bf16* hnb    = (__bf16*)alloc((size_t)Mrows * Cv * 2);       // 16 MB
  float*  stats  = (float*)alloc((size_t)Bv * Hh * 2 * 4);

  const long nElem = (long)Mrows * Cv;

  // 1) weight conversions
  k_cvt<<<(unsigned)(((long)Ngates * Cv + 255) / 256), 256, 0, stream>>>(
      wx_w, wxw_b, (long)Ngates * Cv);
  k_cvt<<<(unsigned)(((long)Cv * Cv + 255) / 256), 256, 0, stream>>>(
      out_w, outw_b, (long)Cv * Cv);

  // 2) depthwise causal conv + SiLU -> bf16
  k_conv<<<(unsigned)((nElem + 255) / 256), 256, 0, stream>>>(x, conv_w, conv_b,
                                                              xcb);

  // 3) gates GEMM (8192 x 4096 x 1024), tanh fused on z columns
  {
    dim3 grid(Ngates / 128, Mrows / 128);
    k_gemm<1, 0><<<grid, 256, 0, stream>>>(xcb, wxw_b, wx_b, nullptr, gates,
                                           Mrows, Ngates, Cv);
  }

  // 4) sLSTM scan + fused GroupNorm statistics (16 blocks, serial in T)
  k_scan<<<Bv * Hh, 256, 0, stream>>>(gates, hbuf, stats);

  // 5) GroupNorm apply -> bf16
  k_gnapply<<<(unsigned)((nElem + 255) / 256), 256, 0, stream>>>(hbuf, stats,
                                                                 gn_w, gn_b,
                                                                 hnb);

  // 6) output GEMM (8192 x 1024 x 1024) + out_b + residual x
  {
    dim3 grid(Cv / 128, Mrows / 128);
    k_gemm<0, 1><<<grid, 256, 0, stream>>>(hnb, outw_b, out_b, x, out, Mrows,
                                           Cv, Cv);
  }
}
